// ImprovedActorCriticNetwork_83373905150670
// MI455X (gfx1250) — matrix operand order repaired and checked by reading, not compiled, not run on hardware
//
#include <hip/hip_runtime.h>

// ---------------------------------------------------------------------------
// Problem constants (match the reference)
// ---------------------------------------------------------------------------
#define N_UAVS   50
#define N_MISS   400
#define NN       (N_UAVS * N_MISS)   // 20000 nodes
#define EE       160000              // edges
#define HEADS    8
#define HC       64
#define HID      512
#define EMB      64
#define COMB     67
#define AH       128
#define CH       128

typedef unsigned short u16;
typedef __bf16 bf16_t;
typedef bf16_t bfvec16 __attribute__((ext_vector_type(16)));
typedef float  fvec8   __attribute__((ext_vector_type(8)));

static __device__ __forceinline__ u16 f32_to_bf16(float f) {
  unsigned int u = __float_as_uint(f);
  u += 0x7fffu + ((u >> 16) & 1u);          // round-to-nearest-even
  return (u16)(u >> 16);
}

static __device__ __forceinline__ void atomic_max_float(float* addr, float v) {
  if (v >= 0.f) atomicMax((int*)addr, __float_as_int(v));
  else          atomicMin((unsigned int*)addr, __float_as_uint(v));
}

// ---------------------------------------------------------------------------
// WMMA bf16 GEMM:  Y[rows,M] = X[rows,Kpad](bf16) * W + bias
// Wt is the transposed weight, bf16, row-major [M, Kpad] (K zero-padded).
// One wave computes a 16x64 tile; rows % 16 == 0, M % 64 == 0, Kpad % 32 == 0.
// ---------------------------------------------------------------------------
__global__ __launch_bounds__(256) void wmma_gemm_bf16(
    const u16* __restrict__ X, const u16* __restrict__ Wt,
    const float* __restrict__ bias, float* __restrict__ Y,
    int rows, int Kpad, int M)
{
  const int wave = (int)((blockIdx.x * blockDim.x + threadIdx.x) >> 5);
  const int lane = (int)(threadIdx.x & 31);
  const int colGroups = M >> 6;
  const int numTiles  = (rows >> 4) * colGroups;
  if (wave >= numTiles) return;                       // uniform per wave
  const int rt   = wave / colGroups;
  const int cg   = wave - rt * colGroups;
  const int row0 = rt << 4;
  const int col0 = cg << 6;
  const int lhi  = lane >> 4;                         // 0 or 1
  const int lm   = lane & 15;

  fvec8 acc0 = {}, acc1 = {}, acc2 = {}, acc3 = {};

  // A fragment: lane holds row (row0+lm); K = k + 16*lhi + 0..15.
  const u16* xrow = X + (size_t)(row0 + lm) * Kpad + (lhi << 4);
  // B fragment: lane holds output col (col0+t*16+lm); K = k + 16*lhi + 0..15.
  const u16* w0 = Wt + (size_t)(col0 + lm) * Kpad + (lhi << 4);
  const u16* w1 = w0 + (size_t)16 * Kpad;
  const u16* w2 = w0 + (size_t)32 * Kpad;
  const u16* w3 = w0 + (size_t)48 * Kpad;

  for (int k = 0; k < Kpad; k += 32) {
    union { bfvec16 v; uint4 u[2]; } a, b0, b1, b2, b3;
    a.u[0]  = *(const uint4*)(xrow + k);
    a.u[1]  = *(const uint4*)(xrow + k + 8);
    b0.u[0] = *(const uint4*)(w0 + k);
    b0.u[1] = *(const uint4*)(w0 + k + 8);
    b1.u[0] = *(const uint4*)(w1 + k);
    b1.u[1] = *(const uint4*)(w1 + k + 8);
    b2.u[0] = *(const uint4*)(w2 + k);
    b2.u[1] = *(const uint4*)(w2 + k + 8);
    b3.u[0] = *(const uint4*)(w3 + k);
    b3.u[1] = *(const uint4*)(w3 + k + 8);
    acc0 = __builtin_amdgcn_wmma_f32_16x16x32_bf16(false, a.v, false, b0.v, (short)0, acc0, false, false);
    acc1 = __builtin_amdgcn_wmma_f32_16x16x32_bf16(false, a.v, false, b1.v, (short)0, acc1, false, false);
    acc2 = __builtin_amdgcn_wmma_f32_16x16x32_bf16(false, a.v, false, b2.v, (short)0, acc2, false, false);
    acc3 = __builtin_amdgcn_wmma_f32_16x16x32_bf16(false, a.v, false, b3.v, (short)0, acc3, false, false);
  }

  const float bv0 = bias[col0 + lm];
  const float bv1 = bias[col0 + 16 + lm];
  const float bv2 = bias[col0 + 32 + lm];
  const float bv3 = bias[col0 + 48 + lm];
#pragma unroll
  for (int r = 0; r < 8; ++r) {
    float* yrow = Y + (size_t)(row0 + (lhi << 3) + r) * M + col0 + lm;
    yrow[0]  = acc0[r] + bv0;
    yrow[16] = acc1[r] + bv1;
    yrow[32] = acc2[r] + bv2;
    yrow[48] = acc3[r] + bv3;
  }
}

// ---------------------------------------------------------------------------
// Node features -> bf16, padded to 32 channels (layer 0 input)
// ---------------------------------------------------------------------------
__global__ void featurize_kernel(const float* __restrict__ mc,
                                 const int* __restrict__ mask,
                                 const float* __restrict__ speeds,
                                 const float* __restrict__ dist,
                                 const float* __restrict__ ttg,
                                 u16* __restrict__ xb)
{
  int n = blockIdx.x * blockDim.x + threadIdx.x;
  if (n >= NN) return;
  int u = n / N_MISS, mi = n - u * N_MISS;
  float f[6];
  f[0] = mc[mi * 2 + 0];
  f[1] = mc[mi * 2 + 1];
  f[2] = (float)mask[n];
  f[3] = speeds[u];
  f[4] = dist[n];
  f[5] = ttg[n];
  u16* row = xb + (size_t)n * 32;
#pragma unroll
  for (int j = 0; j < 6; ++j) row[j] = f32_to_bf16(f[j]);
#pragma unroll
  for (int j = 6; j < 32; ++j) row[j] = 0;
}

// W[K,M] f32 -> Wt[M,Kpad] bf16 (transposed, K zero-padded)
__global__ void convert_wt_kernel(const float* __restrict__ W, u16* __restrict__ Wt,
                                  int K, int M, int Kpad)
{
  int i = blockIdx.x * blockDim.x + threadIdx.x;
  if (i >= M * Kpad) return;
  int m = i / Kpad, k = i - m * Kpad;
  float v = (k < K) ? W[(size_t)k * M + m] : 0.f;
  Wt[i] = f32_to_bf16(v);
}

// ---------------------------------------------------------------------------
// Edge attention
// ---------------------------------------------------------------------------
__global__ __launch_bounds__(256) void edge_logits_kernel(
    const int* __restrict__ ei, const float* __restrict__ Q,
    const float* __restrict__ K, float* __restrict__ logits)
{
  int wave = (int)((blockIdx.x * blockDim.x + threadIdx.x) >> 5);
  int lane = (int)(threadIdx.x & 31);
  if (wave >= EE) return;
  int src = ei[wave];
  int dst = ei[EE + wave];
  const float4* qp = (const float4*)(Q + (size_t)dst * HID) + lane * 4;
  const float4* kp = (const float4*)(K + (size_t)src * HID) + lane * 4;
  float s = 0.f;
#pragma unroll
  for (int i = 0; i < 4; ++i) {
    float4 q = qp[i], k = kp[i];
    s += q.x * k.x + q.y * k.y + q.z * k.z + q.w * k.w;
  }
  s += __shfl_xor(s, 1, 32);
  s += __shfl_xor(s, 2, 32);
  if ((lane & 3) == 0) logits[(size_t)wave * HEADS + (lane >> 2)] = s * 0.125f; // 1/sqrt(64)
}

__global__ void attn_init_kernel(float* __restrict__ m, float* __restrict__ denom,
                                 float* __restrict__ agg)
{
  int i = blockIdx.x * blockDim.x + threadIdx.x;
  if (i < NN * HID) agg[i] = 0.f;
  if (i < NN * HEADS) { m[i] = -__builtin_inff(); denom[i] = 0.f; }
}

__global__ void seg_max_kernel(const int* __restrict__ ei, const float* __restrict__ logits,
                               float* __restrict__ m)
{
  int i = blockIdx.x * blockDim.x + threadIdx.x;
  if (i >= EE * HEADS) return;
  int e = i >> 3, h = i & 7;
  int dst = ei[EE + e];
  atomic_max_float(&m[dst * HEADS + h], logits[i]);
}

__global__ __launch_bounds__(256) void edge_accum_kernel(
    const int* __restrict__ ei, const float* __restrict__ logits,
    const float* __restrict__ m, const float* __restrict__ V,
    float* __restrict__ denom, float* __restrict__ agg)
{
  int wave = (int)((blockIdx.x * blockDim.x + threadIdx.x) >> 5);
  int lane = (int)(threadIdx.x & 31);
  if (wave >= EE) return;
  int src = ei[wave];
  int dst = ei[EE + wave];
  int head = lane >> 2;
  float lg = logits[(size_t)wave * HEADS + head];
  float mm = m[dst * HEADS + head];
  float a = __expf(lg - mm);
  if ((lane & 3) == 0) atomicAdd(&denom[dst * HEADS + head], a);
  const float* vs = V + (size_t)src * HID + lane * 16;
  float* ag = agg + (size_t)dst * HID + lane * 16;
#pragma unroll
  for (int i = 0; i < 16; ++i) atomicAdd(&ag[i], a * vs[i]);
}

// x_next = relu(agg/denom + skip)  -> bf16
__global__ void finalize_kernel(const float* __restrict__ agg,
                                const float* __restrict__ denom,
                                const float* __restrict__ skip,
                                u16* __restrict__ xb)
{
  int i = blockIdx.x * blockDim.x + threadIdx.x;
  if (i >= NN * HID) return;
  int n = i >> 9, c = i & (HID - 1);
  float val = agg[i] / denom[n * HEADS + (c >> 6)] + skip[i];
  val = fmaxf(val, 0.f);
  xb[i] = f32_to_bf16(val);
}

// ---------------------------------------------------------------------------
// Mean-pool per UAV and build combined[U,67]
// ---------------------------------------------------------------------------
__global__ void pool_combine_kernel(const float* __restrict__ emb,
                                    const float* __restrict__ uavs_info,
                                    const float* __restrict__ speeds,
                                    float* __restrict__ combined)
{
  int u = blockIdx.x;      // 50 blocks
  int c = threadIdx.x;     // 64 threads
  float s = 0.f;
  const float* base = emb + (size_t)u * N_MISS * EMB + c;
  for (int mi = 0; mi < N_MISS; ++mi) s += base[(size_t)mi * EMB];
  combined[u * COMB + 2 + c] = s * (1.f / N_MISS);
  if (c == 0) {
    combined[u * COMB + 0]  = uavs_info[u * 2 + 0];
    combined[u * COMB + 1]  = uavs_info[u * 2 + 1];
    combined[u * COMB + 66] = speeds[u];
  }
}

// Tiny fp32 linear for actor/critic heads: Y[r,m] = b[m] + X[r,:] . W[:,m]
__global__ void lin_small_kernel(const float* __restrict__ X, const float* __restrict__ W,
                                 const float* __restrict__ b, float* __restrict__ Y,
                                 int rows, int K, int M, int relu)
{
  int i = blockIdx.x * blockDim.x + threadIdx.x;
  if (i >= rows * M) return;
  int r = i / M, m = i - r * M;
  float s = b[m];
  for (int k = 0; k < K; ++k) s += X[(size_t)r * K + k] * W[(size_t)k * M + m];
  Y[i] = relu ? fmaxf(s, 0.f) : s;
}

// ---------------------------------------------------------------------------
// Host launch
// ---------------------------------------------------------------------------
extern "C" void kernel_launch(void* const* d_in, const int* in_sizes, int n_in,
                              void* d_out, int out_size, void* d_ws, size_t ws_size,
                              hipStream_t stream)
{
  (void)in_sizes; (void)n_in; (void)out_size; (void)ws_size;

  const float* mission_coords = (const float*)d_in[0];
  const int*   edge_index     = (const int*)d_in[1];
  const float* uavs_info      = (const float*)d_in[3];
  const int*   action_mask    = (const int*)d_in[4];
  const float* speeds         = (const float*)d_in[5];
  const float* dist           = (const float*)d_in[6];
  const float* ttg            = (const float*)d_in[7];

  // params pytree leaf order: dict keys sorted; lists in order; {b,w} per linear.
  // actor: 8..13, critic: 14..19, gnn_out: 20(b),21(w),
  // layers[l]: base=22+8*l -> k.b,k.w,q.b,q.w,skip.b,skip.w,v.b,v.w
  const float* actor_b[3] = { (const float*)d_in[8],  (const float*)d_in[10], (const float*)d_in[12] };
  const float* actor_w[3] = { (const float*)d_in[9],  (const float*)d_in[11], (const float*)d_in[13] };
  const float* critic_b[3]= { (const float*)d_in[14], (const float*)d_in[16], (const float*)d_in[18] };
  const float* critic_w[3]= { (const float*)d_in[15], (const float*)d_in[17], (const float*)d_in[19] };
  const float* gnn_b = (const float*)d_in[20];
  const float* gnn_w = (const float*)d_in[21];
  const float *lay_kb[4], *lay_kw[4], *lay_qb[4], *lay_qw[4], *lay_sb[4], *lay_sw[4], *lay_vb[4], *lay_vw[4];
  for (int l = 0; l < 4; ++l) {
    int base = 22 + 8 * l;
    lay_kb[l] = (const float*)d_in[base + 0];
    lay_kw[l] = (const float*)d_in[base + 1];
    lay_qb[l] = (const float*)d_in[base + 2];
    lay_qw[l] = (const float*)d_in[base + 3];
    lay_sb[l] = (const float*)d_in[base + 4];
    lay_sw[l] = (const float*)d_in[base + 5];
    lay_vb[l] = (const float*)d_in[base + 6];
    lay_vw[l] = (const float*)d_in[base + 7];
  }

  // ---- workspace layout ----
  char* ws = (char*)d_ws;
  size_t off = 0;
  auto alloc = [&](size_t bytes) -> void* {
    void* p = ws + off;
    off = (off + bytes + 255) & ~(size_t)255;
    return p;
  };
  u16*   xb     = (u16*)  alloc((size_t)NN * HID * sizeof(u16));   // bf16 node feats (layer0 uses [NN,32])
  float* Qb     = (float*)alloc((size_t)NN * HID * sizeof(float)); // Q, later skip / emb
  float* Kb     = (float*)alloc((size_t)NN * HID * sizeof(float)); // K, later agg
  float* Vb     = (float*)alloc((size_t)NN * HID * sizeof(float));
  float* logits = (float*)alloc((size_t)EE * HEADS * sizeof(float));
  float* mbuf   = (float*)alloc((size_t)NN * HEADS * sizeof(float));
  float* dbuf   = (float*)alloc((size_t)NN * HEADS * sizeof(float));
  u16* wt_q[4]; u16* wt_k[4]; u16* wt_v[4]; u16* wt_s[4];
  for (int l = 0; l < 4; ++l) {
    int Kpad = (l == 0) ? 32 : HID;
    wt_k[l] = (u16*)alloc((size_t)HID * Kpad * sizeof(u16));
    wt_q[l] = (u16*)alloc((size_t)HID * Kpad * sizeof(u16));
    wt_s[l] = (u16*)alloc((size_t)HID * Kpad * sizeof(u16));
    wt_v[l] = (u16*)alloc((size_t)HID * Kpad * sizeof(u16));
  }
  u16*   wt_g   = (u16*)  alloc((size_t)EMB * HID * sizeof(u16));
  float* comb   = (float*)alloc((size_t)N_UAVS * COMB * sizeof(float));
  float* h1     = (float*)alloc((size_t)N_UAVS * AH * sizeof(float));
  float* h2     = (float*)alloc((size_t)N_UAVS * AH * sizeof(float));

  // ---- convert weights to transposed/padded bf16 ----
  for (int l = 0; l < 4; ++l) {
    int Kin = (l == 0) ? 6 : HID;
    int Kpad = (l == 0) ? 32 : HID;
    int tot = HID * Kpad, blk = (tot + 255) / 256;
    convert_wt_kernel<<<blk, 256, 0, stream>>>(lay_kw[l], wt_k[l], Kin, HID, Kpad);
    convert_wt_kernel<<<blk, 256, 0, stream>>>(lay_qw[l], wt_q[l], Kin, HID, Kpad);
    convert_wt_kernel<<<blk, 256, 0, stream>>>(lay_sw[l], wt_s[l], Kin, HID, Kpad);
    convert_wt_kernel<<<blk, 256, 0, stream>>>(lay_vw[l], wt_v[l], Kin, HID, Kpad);
  }
  {
    int tot = EMB * HID, blk = (tot + 255) / 256;
    convert_wt_kernel<<<blk, 256, 0, stream>>>(gnn_w, wt_g, HID, EMB, HID);
  }

  // ---- node features ----
  featurize_kernel<<<(NN + 255) / 256, 256, 0, stream>>>(mission_coords, action_mask,
                                                         speeds, dist, ttg, xb);

  const int gemmTiles512 = (NN / 16) * (HID / 64);          // 10000 waves
  const int gemmBlk512   = (gemmTiles512 + 7) / 8;          // 8 waves/block
  const int nodeElems    = NN * HID;
  const int nodeBlk      = (nodeElems + 255) / 256;
  const int edgeWaveBlk  = (EE + 7) / 8;
  const int segBlk       = (EE * HEADS + 255) / 256;

  // ---- 4 TransformerConv layers ----
  for (int l = 0; l < 4; ++l) {
    int Kpad = (l == 0) ? 32 : HID;
    wmma_gemm_bf16<<<gemmBlk512, 256, 0, stream>>>(xb, wt_q[l], lay_qb[l], Qb, NN, Kpad, HID);
    wmma_gemm_bf16<<<gemmBlk512, 256, 0, stream>>>(xb, wt_k[l], lay_kb[l], Kb, NN, Kpad, HID);
    wmma_gemm_bf16<<<gemmBlk512, 256, 0, stream>>>(xb, wt_v[l], lay_vb[l], Vb, NN, Kpad, HID);

    edge_logits_kernel<<<edgeWaveBlk, 256, 0, stream>>>(edge_index, Qb, Kb, logits);

    // Q,K consumed: reuse Qb as skip, Kb as aggregation buffer
    wmma_gemm_bf16<<<gemmBlk512, 256, 0, stream>>>(xb, wt_s[l], lay_sb[l], Qb, NN, Kpad, HID);

    attn_init_kernel<<<nodeBlk, 256, 0, stream>>>(mbuf, dbuf, Kb);
    seg_max_kernel<<<segBlk, 256, 0, stream>>>(edge_index, logits, mbuf);
    edge_accum_kernel<<<edgeWaveBlk, 256, 0, stream>>>(edge_index, logits, mbuf, Vb, dbuf, Kb);
    finalize_kernel<<<nodeBlk, 256, 0, stream>>>(Kb, dbuf, Qb, xb);
  }

  // ---- gnn_out: [NN,512] -> [NN,64] (emb into Qb) ----
  {
    int tiles = (NN / 16) * (EMB / 64);
    int blk = (tiles + 7) / 8;
    wmma_gemm_bf16<<<blk, 256, 0, stream>>>(xb, wt_g, gnn_b, Qb, NN, HID, EMB);
  }

  // ---- pool + combined ----
  pool_combine_kernel<<<N_UAVS, EMB, 0, stream>>>(Qb, uavs_info, speeds, comb);

  float* out_logits = (float*)d_out;            // [50,400]
  float* out_values = (float*)d_out + N_UAVS * N_MISS;  // [50]

  // actor
  lin_small_kernel<<<(N_UAVS * AH + 255) / 256, 256, 0, stream>>>(comb, actor_w[0], actor_b[0], h1, N_UAVS, COMB, AH, 1);
  lin_small_kernel<<<(N_UAVS * AH + 255) / 256, 256, 0, stream>>>(h1, actor_w[1], actor_b[1], h2, N_UAVS, AH, AH, 1);
  lin_small_kernel<<<(N_UAVS * N_MISS + 255) / 256, 256, 0, stream>>>(h2, actor_w[2], actor_b[2], out_logits, N_UAVS, AH, N_MISS, 0);
  // critic (reuse h1/h2)
  lin_small_kernel<<<(N_UAVS * CH + 255) / 256, 256, 0, stream>>>(comb, critic_w[0], critic_b[0], h1, N_UAVS, COMB, CH, 1);
  lin_small_kernel<<<(N_UAVS * CH + 255) / 256, 256, 0, stream>>>(h1, critic_w[1], critic_b[1], h2, N_UAVS, CH, CH, 1);
  lin_small_kernel<<<(N_UAVS + 255) / 256, 256, 0, stream>>>(h2, critic_w[2], critic_b[2], out_values, N_UAVS, CH, 1, 0);
}
